// RNN_41635412967547
// MI455X (gfx1250) — matrix operand order repaired and checked by reading, not compiled, not run on hardware
//
#include <hip/hip_runtime.h>
#include <hip/hip_bf16.h>
#include <math.h>

// CDNA5 wave32 WMMA types
typedef __attribute__((ext_vector_type(16))) __bf16 v16bf;
typedef __attribute__((ext_vector_type(8)))  float  v8f;

#define T_STEPS 512
#define BATCH   256
#define IN_SZ   1024
#define HID_SZ  1024
#define LD      1024   // leading dimension of every matrix in this problem

// Branch-free tanh: copysign(1 - 2/(exp(2|x|)+1), x).
// exp overflow -> +inf -> rcp -> 0 -> result 1 (correct saturation).
__device__ __forceinline__ float fast_tanh(float x) {
    float ax = __builtin_fabsf(x);
    float e  = __expf(2.0f * ax);                       // v_exp_f32 path
    float r  = 1.0f - 2.0f * __builtin_amdgcn_rcpf(e + 1.0f);
    return __builtin_copysignf(r, x);
}

// Convert 8 consecutive fp32 (two float4 loads) into bf16 elems [off..off+7].
__device__ __forceinline__ void cvt8(const float* __restrict__ p, v16bf& a, int off) {
    float4 f0 = ((const float4*)p)[0];
    float4 f1 = ((const float4*)p)[1];
    a[off + 0] = (__bf16)f0.x; a[off + 1] = (__bf16)f0.y;
    a[off + 2] = (__bf16)f0.z; a[off + 3] = (__bf16)f0.w;
    a[off + 4] = (__bf16)f1.x; a[off + 5] = (__bf16)f1.y;
    a[off + 6] = (__bf16)f1.z; a[off + 7] = (__bf16)f1.w;
}

// ---------------------------------------------------------------------------
// Fused GEMM:  Out[m,n] = f( sum_k A[m,k]*W[n,k] [+ bias[n]] [+ Cin[m,n]] )
//   BIAS: add bias[n].  ADDTANH: add Cin[m,n] then tanh (Out may alias Cin).
//   doK == 0 -> skip GEMM (h_{-1} = 0 at t==0).
// A: fp32 [M,1024] row-major, converted to bf16 on the fly.
// W: bf16 [N,1024] row-major (pre-converted nn.Linear weights).
//
// Fragment layouts used here (contiguous K order, same in A and B; C/D as ISA):
//  A 16x32: lane -> row = lane&15, K = (lane>>4)*16 .. +15 (contiguous)
//  B 32x16: lane -> col = lane&15, K = (lane>>4)*16 .. +15 (contiguous in W row)
//  C/D 16x16: elem v -> m = v + 8*(lane>>4), n = lane&15
// ---------------------------------------------------------------------------
template<int WAVES_M, int WAVES_N, int WM, int WN, bool BIAS, bool ADDTANH>
__global__ __launch_bounds__(32 * WAVES_M * WAVES_N)
void rnn_wmma_gemm(const float* __restrict__ A, const __bf16* __restrict__ W,
                   const float* __restrict__ bias, const float* __restrict__ Cin,
                   float* __restrict__ Out, int doK)
{
    const int wave = threadIdx.x >> 5;
    const int wm   = wave % WAVES_M;
    const int wn   = wave / WAVES_M;
    const int m0   = blockIdx.y * (WAVES_M * WM * 16) + wm * (WM * 16);
    const int n0   = blockIdx.x * (WAVES_N * WN * 16) + wn * (WN * 16);

    const int lane = threadIdx.x & 31;
    const int lr   = lane & 15;            // row (A) / col (B) within tile
    const int k0a  = (lane >> 4) << 4;     // A K-base: 0 or 16
    const int ksb  = (lane >> 4) << 4;     // B K-base: 0 or 16

    v8f acc[WM][WN];
    const v8f vzero = {};
#pragma unroll
    for (int i = 0; i < WM; ++i)
#pragma unroll
        for (int j = 0; j < WN; ++j)
            acc[i][j] = vzero;

    if (doK) {
        // 32-bit base indices; tile strides become immediate offsets.
        const int baseA = (m0 + lr) * LD + k0a;
        const int baseB = (n0 + lr) * LD + ksb;
        for (int kc = 0; kc < LD; kc += 32) {
            v16bf af[WM];
            v16bf bf[WN];
#pragma unroll
            for (int i = 0; i < WM; ++i) {
                cvt8(&A[baseA + i * (16 * LD) + kc],      af[i], 0);
                cvt8(&A[baseA + i * (16 * LD) + kc + 8], af[i], 8);
            }
#pragma unroll
            for (int j = 0; j < WN; ++j)
                bf[j] = *(const v16bf*)&W[baseB + j * (16 * LD) + kc];
#pragma unroll
            for (int i = 0; i < WM; ++i)
#pragma unroll
                for (int j = 0; j < WN; ++j)
                    acc[i][j] = __builtin_amdgcn_wmma_f32_16x16x32_bf16(
                        false, af[i], false, bf[j], (short)0, acc[i][j],
                        false, false);
        }
    }

    // Epilogue: branch-free; column stride LD*4 B fits the immediate offset.
    const int nn = lane & 15;
    const int mo = (lane >> 4) << 3;
#pragma unroll
    for (int i = 0; i < WM; ++i) {
#pragma unroll
        for (int j = 0; j < WN; ++j) {
            const int ng   = n0 + j * 16 + nn;
            const float bv = BIAS ? bias[ng] : 0.0f;
            const int base = (m0 + i * 16 + mo) * LD + ng;
#pragma unroll
            for (int v = 0; v < 8; ++v) {
                float rr = acc[i][j][v] + bv;
                if (ADDTANH) rr = fast_tanh(rr + Cin[base + v * LD]);
                Out[base + v * LD] = rr;
            }
        }
    }
}

// ---------------------------------------------------------------------------
// One-time prep: Wx,Wh fp32 -> bf16 in workspace; fold bx+bh into one bias.
// ---------------------------------------------------------------------------
__global__ void rnn_prep_kernel(const float* __restrict__ Wx, const float* __restrict__ Wh,
                                const float* __restrict__ bx, const float* __restrict__ bh,
                                __bf16* __restrict__ wxb, __bf16* __restrict__ whb,
                                float* __restrict__ bxh)
{
    const int i = blockIdx.x * blockDim.x + threadIdx.x;
    if (i < HID_SZ * IN_SZ) {
        wxb[i] = (__bf16)Wx[i];
        whb[i] = (__bf16)Wh[i];
    }
    if (i < HID_SZ) bxh[i] = bx[i] + bh[i];
}

extern "C" void kernel_launch(void* const* d_in, const int* in_sizes, int n_in,
                              void* d_out, int out_size, void* d_ws, size_t ws_size,
                              hipStream_t stream)
{
    const float* x  = (const float*)d_in[0];   // [T,B,I]
    const float* Wx = (const float*)d_in[1];   // [H,I]
    const float* bx = (const float*)d_in[2];   // [H]
    const float* Wh = (const float*)d_in[3];   // [H,H]
    const float* bh = (const float*)d_in[4];   // [H]

    float* out     = (float*)d_out;
    float* h_final = out;                                  // [B,H]
    float* h_all   = out + (size_t)BATCH * HID_SZ;         // [T,B,H]

    // Workspace layout: Wx(bf16) | Wh(bf16) | bxh(f32)  (~4.2 MB)
    __bf16* wxb = (__bf16*)d_ws;
    __bf16* whb = wxb + (size_t)HID_SZ * IN_SZ;
    float*  bxh = (float*)(whb + (size_t)HID_SZ * HID_SZ);

    // 1) weight conversion + bias folding
    rnn_prep_kernel<<<(HID_SZ * IN_SZ + 255) / 256, 256, 0, stream>>>(
        Wx, Wh, bx, bh, wxb, whb, bxh);

    // 2) input projection for all timesteps: xp = x·Wxᵀ + (bx+bh), into h_all
    //    M = T*B = 131072, N = 1024, K = 1024.  Block tile 128x128.
    {
        dim3 grid(HID_SZ / 128, (T_STEPS * BATCH) / 128);
        rnn_wmma_gemm<4, 2, 2, 4, true, false><<<grid, 256, 0, stream>>>(
            x, wxb, bxh, nullptr, h_all, 1);
    }

    // 3) recurrence: h_t = tanh(xp_t + h_{t-1}·Whᵀ), in place in h_all[t]
    //    M = 256, N = 1024, K = 1024. Block tile 64x256, grid 4x4.
    {
        dim3 grid(HID_SZ / 256, BATCH / 64);
        for (int t = 0; t < T_STEPS; ++t) {
            float* ht = h_all + (size_t)t * BATCH * HID_SZ;          // xp_t in, h_t out
            const float* hprev = (t == 0) ? ht                       // unused when doK=0
                                          : (h_all + (size_t)(t - 1) * BATCH * HID_SZ);
            rnn_wmma_gemm<2, 4, 2, 4, false, true><<<grid, 256, 0, stream>>>(
                hprev, whb, nullptr, /*Cin=*/ht, /*Out=*/ht, t == 0 ? 0 : 1);
        }
    }

    // 4) h_final = h_all[T-1]
    (void)hipMemcpyAsync(h_final, h_all + (size_t)(T_STEPS - 1) * BATCH * HID_SZ,
                         (size_t)BATCH * HID_SZ * sizeof(float),
                         hipMemcpyDeviceToDevice, stream);
}
